// ttconv_21002390077454
// MI455X (gfx1250) — matrix operand order repaired and checked by reading, not compiled, not run on hardware
//
#include <hip/hip_runtime.h>

typedef __attribute__((ext_vector_type(16))) __bf16 v16bf;
typedef __attribute__((ext_vector_type(8)))  __bf16 v8bf;
typedef __attribute__((ext_vector_type(8)))  float  v8f;
typedef __attribute__((ext_vector_type(4)))  unsigned int u32x4;
typedef __attribute__((ext_vector_type(8)))  unsigned int u32x8;

#define D_OUT 256
#define C_IN  128
#define HW    64
#define NB    32
#define R1    8
#define R2    4
#define KSZ   3

#define KC      32            // K (channels) per chunk = one bf16-WMMA K
#define NCHUNK  36            // 9 taps * (128/32)
#define AS_H    40            // bf16 per A row (32 + 8 pad) -> 80 B rows
#define BS_H    40            // bf16 per B row (32 + 8 pad) -> 80 B rows

// ---------------- Stage 1: rebuild conv kernel, split into bf16 hi/lo -------
// Wt_hi/Wt_lo layout: [khkw][d][c]   (bf16 each; hi+lo ~= f32 weight)
__global__ __launch_bounds__(128) void tt_build_weights(
    const float* __restrict__ core0,   // [256,128,8]
    const float* __restrict__ core1,   // [256,8,3,4]
    const float* __restrict__ core2,   // [256,4,3]
    __bf16* __restrict__ Wt_hi,        // [9][256][128]
    __bf16* __restrict__ Wt_lo)        // [9][256][128]
{
    const int d = blockIdx.x;
    const int c = threadIdx.x;
    __shared__ float s_c1[R1 * KSZ * R2];   // 96
    __shared__ float s_c2[R2 * KSZ];        // 12
    if (c < R1 * KSZ * R2) s_c1[c] = core1[d * (R1 * KSZ * R2) + c];
    if (c < R2 * KSZ)      s_c2[c] = core2[d * (R2 * KSZ) + c];
    __syncthreads();

    float a[R1];
#pragma unroll
    for (int r = 0; r < R1; ++r) a[r] = core0[(d * C_IN + c) * R1 + r];

#pragma unroll
    for (int kh = 0; kh < KSZ; ++kh) {
        float t[R2];
#pragma unroll
        for (int s = 0; s < R2; ++s) {
            float acc = 0.f;
#pragma unroll
            for (int r = 0; r < R1; ++r)
                acc += a[r] * s_c1[(r * KSZ + kh) * R2 + s];
            t[s] = acc;
        }
#pragma unroll
        for (int kw = 0; kw < KSZ; ++kw) {
            float w = 0.f;
#pragma unroll
            for (int s = 0; s < R2; ++s) w += t[s] * s_c2[s * KSZ + kw];
            const int idx = ((kh * KSZ + kw) * D_OUT + d) * C_IN + c;
            const __bf16 hi = (__bf16)w;
            Wt_hi[idx] = hi;
            Wt_lo[idx] = (__bf16)(w - (float)hi);
        }
    }
}

// ---------------- Stage 2: implicit-GEMM conv via split-bf16 WMMA -----------
// One block per (n,h): out[n, 0..255, h, 0..63].
// GEMM: M = d (256), N = w (64), K = (kh,kw,c) (1152), K-chunks of 32.
// A tiles (weights) are DMA'd by the Tensor Data Mover; B tiles (x patches)
// are gathered + split to bf16 hi/lo by the VALU.
// acc += Ah*Bh + Ah*Bl + Al*Bh   (bf16x3 ~ f32 accuracy)
__global__ __launch_bounds__(256) void tt_conv_wmma(
    const float*  __restrict__ x,      // [32,128,64,64]
    const __bf16* __restrict__ Wt_hi,  // [9][256][128]
    const __bf16* __restrict__ Wt_lo,  // [9][256][128]
    const float*  __restrict__ bias,   // [1]
    float* __restrict__ out)           // [32,256,64,64]
{
    __shared__ __bf16 AsH[D_OUT * AS_H];   // 20480 B (TDM-filled, 80B rows)
    __shared__ __bf16 AsL[D_OUT * AS_H];   // 20480 B (TDM-filled, 80B rows)
    __shared__ __bf16 BsH[HW * BS_H];      //  5120 B
    __shared__ __bf16 BsL[HW * BS_H];      //  5120 B

    const int tid  = threadIdx.x;
    const int n    = blockIdx.x >> 6;
    const int h    = blockIdx.x & 63;
    const int wave = tid >> 5;
    const int lane = tid & 31;
    const int lo   = lane & 15;
    const int hi   = lane >> 4;

    const float bv = bias[0];
    v8f acc[2][4];
#pragma unroll
    for (int mi = 0; mi < 2; ++mi)
#pragma unroll
        for (int nj = 0; nj < 4; ++nj)
            acc[mi][nj] = (v8f){bv, bv, bv, bv, bv, bv, bv, bv};

    // B-fill coords: thread -> (w position, 8-channel group)
    const int bp  = tid >> 2;          // p = w: 0..63
    const int bcq = (tid & 3) << 3;    // channel group start within chunk

    // TDM descriptor group 1 (shared by hi/lo tiles):
    //   data_size=1 (2B), pad_enable, pad_interval=3 (16 dwords = 64B rows),
    //   pad_amount=3 (4 dwords = 16B)  -> LDS row stride 80B == AS_H*2
    //   tensor_dim0=128 (c), tensor_dim1=9*256 (rows), tile=32x256,
    //   tensor_dim0_stride=128
    const u32x8 g1 = {
        (1u << 16) | (1u << 20) | (3u << 22) | (3u << 25),
        ((unsigned)C_IN) << 16,
        ((unsigned)(9 * D_OUT) & 0xffffu) << 16,
        32u << 16,
        256u,
        (unsigned)C_IN,
        0u, 0u
    };
    const u32x4 gz = {0u, 0u, 0u, 0u};
    const unsigned ldsH = (unsigned)(uintptr_t)(&AsH[0]);
    const unsigned ldsL = (unsigned)(uintptr_t)(&AsL[0]);

    for (int t = 0; t < NCHUNK; ++t) {
        const int khkw = t >> 2;
        const int c0   = (t & 3) << 5;          // 0,32,64,96
        const int kh   = khkw / 3;
        const int kw   = khkw - kh * 3;

        // ---- A tiles via Tensor Data Mover (one wave issues; EXEC-agnostic)
        if (wave == 0) {
            const unsigned long long gaH = (unsigned long long)(uintptr_t)
                (Wt_hi + ((size_t)(khkw * D_OUT)) * C_IN + c0);
            const unsigned long long gaL = (unsigned long long)(uintptr_t)
                (Wt_lo + ((size_t)(khkw * D_OUT)) * C_IN + c0);
            const u32x4 g0H = {1u, ldsH, (unsigned)gaH,
                               (unsigned)(gaH >> 32) | (2u << 30)};
            const u32x4 g0L = {1u, ldsL, (unsigned)gaL,
                               (unsigned)(gaL >> 32) | (2u << 30)};
            asm volatile("tensor_load_to_lds %0, %1, %2, %3"
                         :: "s"(g0H), "s"(g1), "s"(gz), "s"(gz) : "memory");
            asm volatile("tensor_load_to_lds %0, %1, %2, %3"
                         :: "s"(g0L), "s"(g1), "s"(gz), "s"(gz) : "memory");
        }

        // ---- fill B tiles (transposed): x[n][c0+cc][h_in][w_in] -> Bs*[p][cc]
        {
            const int h_in = h + kh - 1;
            const int w_in = bp + kw - 1;
            const bool ok = ((unsigned)h_in < (unsigned)HW) &&
                            ((unsigned)w_in < (unsigned)HW);
            const float* src = x + (((size_t)(n * C_IN + c0 + bcq)) * HW +
                                    (ok ? h_in : 0)) * HW + (ok ? w_in : 0);
            v8bf vh, vl;
#pragma unroll
            for (int i = 0; i < 8; ++i) {
                const float v = ok ? src[(size_t)i * HW * HW] : 0.f;
                const __bf16 vhi = (__bf16)v;
                vh[i] = vhi;
                vl[i] = (__bf16)(v - (float)vhi);
            }
            *(v8bf*)&BsH[bp * BS_H + bcq] = vh;
            *(v8bf*)&BsL[bp * BS_H + bcq] = vl;
        }

        // TDM completion is per-wave: issuing wave waits, barrier publishes LDS
        if (wave == 0) __builtin_amdgcn_s_wait_tensorcnt(0);
        __syncthreads();

        // ---- compute: one K=32 bf16 WMMA slice, 3 split products
        const int M0base = wave * 32;
        v16bf aH[2], aL[2];
#pragma unroll
        for (int mi = 0; mi < 2; ++mi) {
            // A frag: elems 0-7 = K[8*hi ..), elems 8-15 = K[16+8*hi ..)
            const int base = (M0base + mi * 16 + lo) * AS_H + 8 * hi;
            v8bf h0 = *(const v8bf*)&AsH[base];
            v8bf h1 = *(const v8bf*)&AsH[base + 16];
            v8bf l0 = *(const v8bf*)&AsL[base];
            v8bf l1 = *(const v8bf*)&AsL[base + 16];
            aH[mi] = __builtin_shufflevector(h0, h1, 0,1,2,3,4,5,6,7,8,9,10,11,12,13,14,15);
            aL[mi] = __builtin_shufflevector(l0, l1, 0,1,2,3,4,5,6,7,8,9,10,11,12,13,14,15);
        }
#pragma unroll
        for (int nj = 0; nj < 4; ++nj) {
            // B frag: elems 0-7 = K[8*hi ..), elems 8-15 = K[16+8*hi ..)
            const int base = (nj * 16 + lo) * BS_H + 8 * hi;
            v8bf h0 = *(const v8bf*)&BsH[base];
            v8bf h1 = *(const v8bf*)&BsH[base + 16];
            v8bf l0 = *(const v8bf*)&BsL[base];
            v8bf l1 = *(const v8bf*)&BsL[base + 16];
            v16bf bH = __builtin_shufflevector(h0, h1, 0,1,2,3,4,5,6,7,8,9,10,11,12,13,14,15);
            v16bf bL = __builtin_shufflevector(l0, l1, 0,1,2,3,4,5,6,7,8,9,10,11,12,13,14,15);
#pragma unroll
            for (int mi = 0; mi < 2; ++mi) {
                acc[mi][nj] = __builtin_amdgcn_wmma_f32_16x16x32_bf16(
                    false, aH[mi], false, bH, (short)0, acc[mi][nj], false, false);
                acc[mi][nj] = __builtin_amdgcn_wmma_f32_16x16x32_bf16(
                    false, aH[mi], false, bL, (short)0, acc[mi][nj], false, false);
                acc[mi][nj] = __builtin_amdgcn_wmma_f32_16x16x32_bf16(
                    false, aL[mi], false, bH, (short)0, acc[mi][nj], false, false);
            }
        }
        __syncthreads();
    }

    // ---- store: C/D layout, VGPR v -> row v (lanes 0-15) / v+8 (lanes 16-31)
#pragma unroll
    for (int mi = 0; mi < 2; ++mi) {
#pragma unroll
        for (int nj = 0; nj < 4; ++nj) {
#pragma unroll
            for (int v = 0; v < 8; ++v) {
                const int d = wave * 32 + mi * 16 + v + 8 * hi;
                const int w = nj * 16 + lo;
                out[(((size_t)(n * D_OUT + d)) * HW + h) * HW + w] = acc[mi][nj][v];
            }
        }
    }
}

extern "C" void kernel_launch(void* const* d_in, const int* in_sizes, int n_in,
                              void* d_out, int out_size, void* d_ws, size_t ws_size,
                              hipStream_t stream) {
    (void)in_sizes; (void)n_in; (void)out_size; (void)ws_size;
    const float* x     = (const float*)d_in[0];
    const float* core0 = (const float*)d_in[1];
    const float* core1 = (const float*)d_in[2];
    const float* core2 = (const float*)d_in[3];
    const float* bias  = (const float*)d_in[4];

    __bf16* Wt_hi = (__bf16*)d_ws;                       // 9*256*128 bf16
    __bf16* Wt_lo = Wt_hi + (size_t)9 * D_OUT * C_IN;    // 9*256*128 bf16
    float* out = (float*)d_out;

    tt_build_weights<<<D_OUT, C_IN, 0, stream>>>(core0, core1, core2, Wt_hi, Wt_lo);
    tt_conv_wmma<<<NB * HW, 256, 0, stream>>>(x, Wt_hi, Wt_lo, bias, out);
}
